// TransformerBach_19181323944190
// MI455X (gfx1250) — matrix operand order repaired and checked by reading, not compiled, not run on hardware
//
#include <hip/hip_runtime.h>
#include <hip/hip_bf16.h>

// ---------------- model dims (match reference) ----------------
#define BB   4
#define EE   256
#define CC   4
#define GG   8
#define VV   96
#define ED   64
#define PP   8
#define DD   512
#define HH   8
#define FFD  2048
#define NEL  4
#define NDL  4
#define SS   (EE*CC)      // 1024
#define DHD  (DD/HH)      // 64
#define BS   (BB*SS)      // 4096

typedef __bf16 bf16_t;
typedef bf16_t bf16x16 __attribute__((ext_vector_type(16)));
typedef bf16_t bf16x8  __attribute__((ext_vector_type(8)));
typedef bf16_t bf16x4  __attribute__((ext_vector_type(4)));
typedef bf16_t bf16x2  __attribute__((ext_vector_type(2)));
typedef float  f32x8   __attribute__((ext_vector_type(8)));

static __device__ __forceinline__ bf16_t bf(float x) { return (bf16_t)x; }

// Packed bf16 pair store (selects to a single v_cvt_pk_bf16_f32 + b32 store).
static __device__ __forceinline__ void store_pk(bf16_t* p, float x, float y) {
  bf16x2 v;
  v[0] = bf(x);
  v[1] = bf(y);
  *(bf16x2*)p = v;
}

static __device__ __forceinline__ f32x8 wmma_bf16(bf16x16 a, bf16x16 b, f32x8 c) {
  return __builtin_amdgcn_wmma_f32_16x16x32_bf16(false, a, false, b, (short)0, c,
                                                 false, false);
}

// Build a 16-elem bf16 fragment from two contiguous 8-elem (16B) runs.
// k order used here (contiguous, not the ISA 7.12.2 table; A and B alike): lane half 0 holds K{0..15},
// half 1 holds K{16..31} -> per lane two contiguous 8-bf16 runs.
static __device__ __forceinline__ bf16x16 frag2(const bf16_t* p0, const bf16_t* p1) {
  bf16x8 lo = *(const bf16x8*)p0;
  bf16x8 hi = *(const bf16x8*)p1;
  bf16x16 a;
  #pragma unroll
  for (int e = 0; e < 8; ++e) { a[e] = lo[e]; a[8 + e] = hi[e]; }
  return a;
}

// Convert 8 contiguous f32 (two float4 / global_load_b128) into frag elems.
static __device__ __forceinline__ void cvt8(bf16x16& dst, int off, const float* p) {
  float4 u = *(const float4*)p;
  float4 w = *(const float4*)(p + 4);
  dst[off + 0] = bf(u.x); dst[off + 1] = bf(u.y);
  dst[off + 2] = bf(u.z); dst[off + 3] = bf(u.w);
  dst[off + 4] = bf(w.x); dst[off + 5] = bf(w.y);
  dst[off + 6] = bf(w.z); dst[off + 7] = bf(w.w);
}

// ---------------- embedding + positional + input projection ----------------
__global__ __launch_bounds__(256)
void embed_kernel(const int* __restrict__ x, const int* __restrict__ mp,
                  const float* __restrict__ etab, const float* __restrict__ chemb,
                  const float* __restrict__ evemb, const float* __restrict__ msk,
                  const float* __restrict__ sos, const float* __restrict__ linW,
                  const float* __restrict__ linb,
                  float* __restrict__ src, float* __restrict__ tgt) {
  __shared__ float fs[ED + 2 * PP];
  __shared__ float ft[ED + 2 * PP];
  int r = blockIdx.x;
  int b = r / SS, s = r % SS;
  int e = s / CC, c = s % CC;
  int t = threadIdx.x;
  if (t < ED + 2 * PP) {
    float vs, vt;
    if (t < ED) {
      int tok = x[(b * EE + e) * CC + c];
      float em = etab[(c * VV + tok) * ED + t];
      vt = em;
      vs = mp[(b * EE + e) * CC + c] ? msk[t] : em;
    } else if (t < ED + PP) {
      float v2 = chemb[c * PP + (t - ED)];
      vt = v2; vs = v2;
    } else {
      float v2 = evemb[(e % GG) * PP + (t - ED - PP)];
      vt = v2; vs = v2;
    }
    fs[t] = vs; ft[t] = vt;
  }
  __syncthreads();
  for (int d = t; d < DD; d += 256) {
    float as = linb[d], at = linb[d];
    for (int k = 0; k < ED + 2 * PP; ++k) {
      float w = linW[k * DD + d];
      as += fs[k] * w;
      at += ft[k] * w;
    }
    src[(size_t)r * DD + d] = as;
    if (s == 0)     tgt[(size_t)(b * SS) * DD + d] = sos[d];   // SOS row
    if (s < SS - 1) tgt[(size_t)(r + 1) * DD + d] = at;        // shift right
  }
}

// ---------------- bf16 WMMA GEMM: C = A(MxK)*W(KxN) + bias (+resid)(+relu) ---
// Block tile 128x64, K-step 32; 8 waves, each computes 32x32 (4 WMMAs).
// A staged row-major, B staged transposed -> all fragment loads are b128.
// N, K are template constants (4 shapes) so row strides constant-fold.
// Register-prefetch pipeline: global loads of step k+1 overlap WMMA of step k.
#define TM 128
#define TN 64
#define TKK 32
#define LDA 40   // 80B rows: 16B aligned, bank offset 20 (conflict-free mod 64)
#define LDK 40

template <bool RELU, bool RESID, int N, int K>
__global__ __launch_bounds__(256)
void gemm_kernel(const float* __restrict__ A, const float* __restrict__ W,
                 const float* __restrict__ bias, const float* __restrict__ R,
                 float* __restrict__ C) {
  __shared__ __align__(16) bf16_t As[TM * LDA];   // [row][k]
  __shared__ __align__(16) bf16_t Bt[TN * LDK];   // [col][k] (transposed)
  int m0 = blockIdx.y * TM, n0 = blockIdx.x * TN;
  int tid = threadIdx.x;
  int wave = tid >> 5, lane = tid & 31;
  int wr = wave >> 1, wc = wave & 1;              // 4x2 waves -> 32x32 each
  int half = lane >> 4, l16 = lane & 15;

  float2 aReg[8];
  float  bR0[4], bR1[4];
  auto prefetch = [&](int k0) {
    #pragma unroll
    for (int i = 0; i < 8; ++i) {                 // A: K-pairs, b64 loads
      int e2 = tid + i * 256;
      int rr = e2 >> 4, q = (e2 & 15) * 2;
      aReg[i] = *(const float2*)&A[(size_t)(m0 + rr) * K + k0 + q];
    }
    #pragma unroll
    for (int i = 0; i < 4; ++i) {                 // B: two coalesced rows
      int e2 = tid + i * 256;
      int n = e2 & 63, p = (e2 >> 6) * 2;
      bR0[i] = W[(size_t)(k0 + p) * N + n0 + n];
      bR1[i] = W[(size_t)(k0 + p + 1) * N + n0 + n];
    }
  };
  auto commit = [&]() {                           // packed-pair b32 LDS stores
    #pragma unroll
    for (int i = 0; i < 8; ++i) {
      int e2 = tid + i * 256;
      int rr = e2 >> 4, q = (e2 & 15) * 2;
      store_pk(&As[rr * LDA + q], aReg[i].x, aReg[i].y);
    }
    #pragma unroll
    for (int i = 0; i < 4; ++i) {
      int e2 = tid + i * 256;
      int n = e2 & 63, p = (e2 >> 6) * 2;
      store_pk(&Bt[n * LDK + p], bR0[i], bR1[i]);
    }
  };

  f32x8 acc00 = {}, acc01 = {}, acc10 = {}, acc11 = {};
  prefetch(0);
  for (int k0 = 0; k0 < K; k0 += TKK) {
    commit();
    __syncthreads();
    if (k0 + TKK < K) prefetch(k0 + TKK);         // overlaps WMMA below
    const bf16_t* ar0 = As + (wr * 32 + l16) * LDA;
    const bf16_t* ar1 = ar0 + 16 * LDA;
    bf16x16 a0 = frag2(ar0 + half * 16, ar0 + half * 16 + 8);
    bf16x16 a1 = frag2(ar1 + half * 16, ar1 + half * 16 + 8);
    const bf16_t* bc0 = Bt + (wc * 32 + l16) * LDK + half * 16;
    const bf16_t* bc1 = Bt + (wc * 32 + 16 + l16) * LDK + half * 16;
    bf16x16 b0 = frag2(bc0, bc0 + 8);
    bf16x16 b1 = frag2(bc1, bc1 + 8);
    acc00 = wmma_bf16(a0, b0, acc00);
    acc01 = wmma_bf16(a0, b1, acc01);
    acc10 = wmma_bf16(a1, b0, acc10);
    acc11 = wmma_bf16(a1, b1, acc11);
    __syncthreads();
  }
  #pragma unroll
  for (int i = 0; i < 2; ++i) {
    f32x8 aL = i ? acc10 : acc00;
    f32x8 aR = i ? acc11 : acc01;
    #pragma unroll
    for (int v = 0; v < 8; ++v) {                 // C: M = v + 8*half, N = l16
      int row = m0 + wr * 32 + i * 16 + v + half * 8;
      int col = n0 + wc * 32 + l16;
      float x0 = aL[v] + bias[col];
      float x1 = aR[v] + bias[col + 16];
      if (RESID) {
        x0 += R[(size_t)row * N + col];
        x1 += R[(size_t)row * N + col + 16];
      }
      if (RELU) { x0 = fmaxf(x0, 0.0f); x1 = fmaxf(x1, 0.0f); }
      C[(size_t)row * N + col] = x0;
      C[(size_t)row * N + col + 16] = x1;
    }
  }
}

// ---------------- y = LN(x)*g + b (residual folded upstream) ----------------
__global__ __launch_bounds__(256)
void ln_kernel(const float* __restrict__ X, const float* __restrict__ g,
               const float* __restrict__ be, float* __restrict__ Y) {
  __shared__ float r1[256];
  __shared__ float r2[256];
  int r = blockIdx.x, t = threadIdx.x;
  float x0 = X[(size_t)r * DD + t];
  float x1 = X[(size_t)r * DD + t + 256];
  r1[t] = x0 + x1;
  r2[t] = x0 * x0 + x1 * x1;
  __syncthreads();
  for (int off = 128; off > 0; off >>= 1) {
    if (t < off) { r1[t] += r1[t + off]; r2[t] += r2[t + off]; }
    __syncthreads();
  }
  float mu = r1[0] * (1.0f / DD);
  float var = r2[0] * (1.0f / DD) - mu * mu;
  float rstd = rsqrtf(var + 1e-5f);
  Y[(size_t)r * DD + t] = (x0 - mu) * rstd * g[t] + be[t];
  Y[(size_t)r * DD + t + 256] = (x1 - mu) * rstd * g[t + 256] + be[t + 256];
}

// ---------------- V transpose: f32 [b][s][h*64+d] -> bf16 VT[b][h][d][s] ----
__global__ __launch_bounds__(256)
void vtrans_kernel(const float* __restrict__ V, bf16_t* __restrict__ VT) {
  __shared__ float tile[64][65];
  int s0 = blockIdx.x * 64, h = blockIdx.y, b = blockIdx.z;
  int t = threadIdx.x;
  int tr = t >> 4, tc = (t & 15) * 4;
  #pragma unroll
  for (int i = 0; i < 4; ++i) {                    // read 64(s) x 64(d)
    int sl = tr + i * 16;
    const float* p = V + ((size_t)(b * SS + s0 + sl)) * DD + h * DHD + tc;
    float4 u = *(const float4*)p;
    tile[sl][tc + 0] = u.x; tile[sl][tc + 1] = u.y;
    tile[sl][tc + 2] = u.z; tile[sl][tc + 3] = u.w;
  }
  __syncthreads();
  bf16_t* vth = VT + ((size_t)(b * HH + h) * DHD) * SS;
  #pragma unroll
  for (int i = 0; i < 4; ++i) {                    // write 64(d) x 64(s)
    int idx = t + i * 256;
    int d = idx >> 4, sb = (idx & 15) * 4;
    bf16x4 o;
    o[0] = bf(tile[sb + 0][d]); o[1] = bf(tile[sb + 1][d]);
    o[2] = bf(tile[sb + 2][d]); o[3] = bf(tile[sb + 3][d]);
    *(bf16x4*)&vth[(size_t)d * SS + s0 + sb] = o;
  }
}

// ---------------- fused attention (self-attn, rel-bias + causal masks) ------
// MODE 0: encoder (allow k >= q). MODE 1: decoder (allow k <= q).
// One wave per (b, h, 16-row q tile); 16x1024 f32 score strip in LDS.
#define SCLD 1044  // mult of 4 (16B align) and bank offset 20 -> conflict-free

template <int MODE>
__global__ __launch_bounds__(32)
void attn_kernel(const float* __restrict__ Q, const float* __restrict__ K,
                 const bf16_t* __restrict__ VT, const float* __restrict__ rel,
                 float* __restrict__ O) {
  __shared__ __align__(16) float sc[16 * SCLD];
  int qbase = blockIdx.x * 16;
  int h = blockIdx.y, b = blockIdx.z;
  int lane = threadIdx.x & 31;
  int half = lane >> 4, l16 = lane & 15;
  const float scale = 0.125f;  // 1/sqrt(DH)
  const float* relh = rel + h * (2 * SS - 1);

  bf16x16 aq0, aq1;            // Q tile frags (DH=64 -> two K=32 chunks)
  {
    const float* qrow = Q + ((size_t)(b * SS + qbase + l16)) * DD + h * DHD;
    cvt8(aq0, 0, qrow + half * 16);
    cvt8(aq0, 8, qrow + half * 16 + 8);
    cvt8(aq1, 0, qrow + 32 + half * 16);
    cvt8(aq1, 8, qrow + 32 + half * 16 + 8);
  }
  for (int j = 0; j < SS / 16; ++j) {   // scores = Q K^T * scale + rel + mask
    int kb = j * 16;
    bool dead = MODE ? (kb > qbase + 15) : (kb + 15 < qbase);
    if (dead) {
      #pragma unroll
      for (int v = 0; v < 8; ++v)
        sc[(v + half * 8) * SCLD + kb + l16] = -INFINITY;
      continue;
    }
    bf16x16 b0, b1;                     // B = K^T: lane = K-row, elem = dh
    const float* krow = K + ((size_t)(b * SS + kb + l16)) * DD + h * DHD;
    cvt8(b0, 0, krow + half * 16);
    cvt8(b0, 8, krow + half * 16 + 8);
    cvt8(b1, 0, krow + 32 + half * 16);
    cvt8(b1, 8, krow + 32 + half * 16 + 8);
    f32x8 c = {};
    c = wmma_bf16(aq0, b0, c);
    c = wmma_bf16(aq1, b1, c);
    int kg = kb + l16;
    int tv = kg - qbase - half * 8;     // allow: MODE ? v>=tv : v<=tv
    int ridx = qbase - kg + SS - 1 + half * 8;  // always in [0, 2S-3]
    #pragma unroll
    for (int v = 0; v < 8; ++v) {       // one cmp + one cndmask per element
      float rb = relh[ridx + v];
      bool ok = MODE ? (v >= tv) : (v <= tv);
      float val = ok ? fmaf(c[v], scale, rb) : -INFINITY;
      sc[(v + half * 8) * SCLD + kg] = val;
    }
  }
  __syncthreads();
  for (int rr = 0; rr < 16; ++rr) {     // row softmax, wave-cooperative
    float* rp = sc + rr * SCLD;
    float mx = -INFINITY;
    for (int i = lane; i < SS; i += 32) mx = fmaxf(mx, rp[i]);
    for (int off = 16; off > 0; off >>= 1) mx = fmaxf(mx, __shfl_xor(mx, off, 32));
    float sum = 0.0f;
    for (int i = lane; i < SS; i += 32) {
      float ev = __expf(rp[i] - mx);
      rp[i] = ev;
      sum += ev;
    }
    for (int off = 16; off > 0; off >>= 1) sum += __shfl_xor(sum, off, 32);
    float rinv = 1.0f / sum;
    for (int i = lane; i < SS; i += 32) rp[i] *= rinv;
  }
  __syncthreads();
  f32x8 o[4] = {};                      // out tile 16 x 64
  int kc_lo = MODE ? 0 : (qbase >> 5);
  int kc_hi = MODE ? ((qbase + 15) >> 5) : (SS / 32 - 1);
  const bf16_t* vth = VT + ((size_t)(b * HH + h) * DHD) * SS;
  for (int kc = kc_lo; kc <= kc_hi; ++kc) {
    bf16x16 a;                          // probabilities: contiguous in LDS
    const float* sp = sc + l16 * SCLD + kc * 32;
    cvt8(a, 0, sp + half * 16);
    cvt8(a, 8, sp + half * 16 + 8);
    #pragma unroll
    for (int cg = 0; cg < 4; ++cg) {    // VT[d][s]: two b128 loads per frag
      const bf16_t* vp = vth + (size_t)(cg * 16 + l16) * SS + kc * 32 + half * 16;
      bf16x16 vb = frag2(vp, vp + 8);
      o[cg] = wmma_bf16(a, vb, o[cg]);
    }
  }
  float* orow = O + ((size_t)(b * SS + qbase)) * DD + h * DHD;
  #pragma unroll
  for (int cg = 0; cg < 4; ++cg)
    #pragma unroll
    for (int v = 0; v < 8; ++v)
      orow[(size_t)(v + half * 8) * DD + cg * 16 + l16] = o[cg][v];
}

// ---------------- per-channel heads + log-softmax + mean NLL ----------------
__global__ __launch_bounds__(128)
void nll_kernel(const float* __restrict__ Xd, const int* __restrict__ tok,
                const float* __restrict__ oW, const float* __restrict__ ob,
                float* __restrict__ out) {
  __shared__ float row[DD];
  __shared__ float red[128];
  int r = blockIdx.x, t = threadIdx.x;
  int c = r % CC;
  for (int i = t; i < DD; i += 128) row[i] = Xd[(size_t)r * DD + i];
  __syncthreads();
  float logit = -INFINITY;
  if (t < VV) {
    float a = ob[c * VV + t];
    for (int d = 0; d < DD; ++d) a += row[d] * oW[((size_t)c * DD + d) * VV + t];
    logit = a;
  }
  red[t] = logit;
  __syncthreads();
  for (int off = 64; off > 0; off >>= 1) {
    if (t < off) red[t] = fmaxf(red[t], red[t + off]);
    __syncthreads();
  }
  float mx = red[0];
  __syncthreads();
  red[t] = (t < VV) ? __expf(logit - mx) : 0.0f;
  __syncthreads();
  for (int off = 64; off > 0; off >>= 1) {
    if (t < off) red[t] += red[t + off];
    __syncthreads();
  }
  float lse = logf(red[0]) + mx;
  if (t == tok[r]) atomicAdd(out, (lse - logit) * (1.0f / (BB * EE * CC)));
}

// ---------------------------------------------------------------------------
extern "C" void kernel_launch(void* const* d_in, const int* in_sizes, int n_in,
                              void* d_out, int out_size, void* d_ws, size_t ws_size,
                              hipStream_t stream) {
  (void)in_sizes; (void)n_in;
  const int* xin = (const int*)d_in[0];
  const int* mp = (const int*)d_in[1];
  const float* etab = (const float*)d_in[2];
  const float* chemb = (const float*)d_in[3];
  const float* evemb = (const float*)d_in[4];
  const float* msk = (const float*)d_in[5];
  const float* sos = (const float*)d_in[6];
  const float* linW = (const float*)d_in[7];
  const float* linb = (const float*)d_in[8];
  const float* outW = (const float*)d_in[9];
  const float* outb = (const float*)d_in[10];
  auto F = [&](int i) { return (const float*)d_in[i]; };

  // Workspace: 7 BSxD f32 buffers + 1 BSxD bf16 (VT); FF hidden overlays Q..AO.
  const size_t BSD = (size_t)BS * DD;
  if (ws_size < 7 * BSD * sizeof(float) + BSD * sizeof(bf16_t)) return;
  float* f = (float*)d_ws;
  float* SRCb = f;               // encoder stream / memory
  float* TGTb = f + 1 * BSD;     // decoder stream
  float* Qb   = f + 2 * BSD;
  float* Kb   = f + 3 * BSD;
  float* Vb   = f + 4 * BSD;
  float* AOb  = f + 5 * BSD;
  float* TMPb = f + 6 * BSD;
  float* FFHb = Qb;              // spans Qb..AOb (4*BSD) during FFN phase
  bf16_t* VTb = (bf16_t*)(f + 7 * BSD);

  hipMemsetAsync(d_out, 0, sizeof(float) * out_size, stream);

  embed_kernel<<<BS, 256, 0, stream>>>(xin, mp, etab, chemb, evemb, msk, sos,
                                       linW, linb, SRCb, TGTb);

  // Four static GEMM shapes (N,K template params -> constant-folded strides).
  dim3 gD(DD / TN, BS / TM), gF(FFD / TN, BS / TM);
  auto gemm_proj = [&](const float* A, const float* W, const float* bias, float* C) {
    gemm_kernel<false, false, DD, DD><<<gD, 256, 0, stream>>>(A, W, bias, nullptr, C);
  };
  auto gemm_res = [&](const float* A, const float* W, const float* bias,
                      const float* R, float* C) {
    gemm_kernel<false, true, DD, DD><<<gD, 256, 0, stream>>>(A, W, bias, R, C);
  };
  auto gemm_ff1 = [&](const float* A, const float* W, const float* bias, float* C) {
    gemm_kernel<true, false, FFD, DD><<<gF, 256, 0, stream>>>(A, W, bias, nullptr, C);
  };
  auto gemm_ff2 = [&](const float* A, const float* W, const float* bias,
                      const float* R, float* C) {
    gemm_kernel<false, true, DD, FFD><<<gD, 256, 0, stream>>>(A, W, bias, R, C);
  };
  auto ln = [&](const float* X, const float* g, const float* be, float* Y) {
    ln_kernel<<<BS, 256, 0, stream>>>(X, g, be, Y);
  };
  dim3 ag(SS / 16, HH, BB);
  dim3 vg(SS / 64, HH, BB);

  // ---- encoder: 11 + 17*l ; sa{Wq,Wk,Wv,Wo,bq,bk,bv,bo,rel} ff{W1,b1,W2,b2,g0,be0,g1,be1}
  for (int l = 0; l < NEL; ++l) {
    int p = 11 + 17 * l;
    gemm_proj(SRCb, F(p + 0), F(p + 4), Qb);
    gemm_proj(SRCb, F(p + 1), F(p + 5), Kb);
    gemm_proj(SRCb, F(p + 2), F(p + 6), Vb);
    vtrans_kernel<<<vg, 256, 0, stream>>>(Vb, VTb);
    attn_kernel<0><<<ag, 32, 0, stream>>>(Qb, Kb, VTb, F(p + 8), AOb);
    gemm_res(AOb, F(p + 3), F(p + 7), SRCb, TMPb);
    ln(TMPb, F(p + 13), F(p + 14), SRCb);
    gemm_ff1(SRCb, F(p + 9), F(p + 10), FFHb);
    gemm_ff2(FFHb, F(p + 11), F(p + 12), SRCb, TMPb);
    ln(TMPb, F(p + 15), F(p + 16), SRCb);
  }
  // ---- decoder: 11 + 17*NEL + 28*l ; sa(+0..8) ca(+9..17) ff(+18..27)
  for (int l = 0; l < NDL; ++l) {
    int p = 11 + 17 * NEL + 28 * l;
    gemm_proj(TGTb, F(p + 0), F(p + 4), Qb);
    gemm_proj(TGTb, F(p + 1), F(p + 5), Kb);
    gemm_proj(TGTb, F(p + 2), F(p + 6), Vb);
    vtrans_kernel<<<vg, 256, 0, stream>>>(Vb, VTb);
    attn_kernel<1><<<ag, 32, 0, stream>>>(Qb, Kb, VTb, F(p + 8), AOb);
    gemm_res(AOb, F(p + 3), F(p + 7), TGTb, TMPb);
    ln(TMPb, F(p + 22), F(p + 23), TGTb);
    // Cross-attention: mem_mask is diagonal-only -> softmax weight is exactly 1
    // at k==q, so out == (mem*Wv + bv)*Wo + bo (diag bias cancels). Two GEMMs.
    gemm_proj(SRCb, F(p + 11), F(p + 15), Vb);          // ca Wv, bv
    gemm_res(Vb, F(p + 12), F(p + 16), TGTb, TMPb);     // ca Wo, bo
    ln(TMPb, F(p + 24), F(p + 25), TGTb);
    gemm_ff1(TGTb, F(p + 18), F(p + 19), FFHb);
    gemm_ff2(FFHb, F(p + 20), F(p + 21), TGTb, TMPb);
    ln(TMPb, F(p + 26), F(p + 27), TGTb);
  }
  nll_kernel<<<BB * EE * CC, 128, 0, stream>>>(TGTb, xin, outW, outb, (float*)d_out);
}
